// Model_10299331576573
// MI455X (gfx1250) — matrix operand order repaired and checked by reading, not compiled, not run on hardware
//
#include <hip/hip_runtime.h>
#include <hip/hip_bf16.h>

// GraphSAGE inference on gfx1250 (MI455X): edge mean-aggregation via coalesced
// float4 gathers + global_atomic_add_f32 scatter (memory-bound stage, ~1.6GB
// edge traffic @ 23.3TB/s), and all GEMMs through v_wmma_f32_16x16x32_f16
// (f32 activations converted to f16 fragments, f32 accumulate).

#define NNODES 100000
#define DF 128   // feature / hidden dim for conv layers
#define LH 256   // decoder hidden

typedef __attribute__((ext_vector_type(16))) _Float16 v16h;
typedef __attribute__((ext_vector_type(8)))  float    v8f;

// ---------------------------------------------------------------------------
// A-fragment: 16x32 (MxK) f16, contiguous k order (same as load_b_frag):
//   lanes 0-15 : M=lane,    elements 0..15 -> K=kk+0..15
//   lanes 16-31: M=lane-16, elements 0..15 -> K=kk+16..31
// rowp points at A[row_base + (lane&15)] * ldk + kk; scale folds 1/cnt in.
__device__ __forceinline__ v16h load_a_frag(const float* __restrict__ rowp,
                                            int lane, float scale) {
  const int klo = (lane >> 4) * 16;
  const float4 f0 = *(const float4*)(rowp + klo + 0);
  const float4 f1 = *(const float4*)(rowp + klo + 4);
  const float4 f2 = *(const float4*)(rowp + klo + 8);
  const float4 f3 = *(const float4*)(rowp + klo + 12);
  v16h a;
  a[0]  = (_Float16)(f0.x * scale); a[1]  = (_Float16)(f0.y * scale);
  a[2]  = (_Float16)(f0.z * scale); a[3]  = (_Float16)(f0.w * scale);
  a[4]  = (_Float16)(f1.x * scale); a[5]  = (_Float16)(f1.y * scale);
  a[6]  = (_Float16)(f1.z * scale); a[7]  = (_Float16)(f1.w * scale);
  a[8]  = (_Float16)(f2.x * scale); a[9]  = (_Float16)(f2.y * scale);
  a[10] = (_Float16)(f2.z * scale); a[11] = (_Float16)(f2.w * scale);
  a[12] = (_Float16)(f3.x * scale); a[13] = (_Float16)(f3.y * scale);
  a[14] = (_Float16)(f3.z * scale); a[15] = (_Float16)(f3.w * scale);
  return a;
}

// B-fragment: 32x16 (KxN) f16. out = A @ W^T, W is [O,K] row-major, so
// B[k][n] = W[o_base+n][k]. Lane holds column N=lane&15, 16 consecutive K
// starting at kk + 16*(lane>>4)  -> four contiguous float4 loads.
__device__ __forceinline__ v16h load_b_frag(const float* __restrict__ W,
                                            int ldk, int o_base, int kk, int lane) {
  const int n  = lane & 15;
  const int kb = (lane >> 4) * 16;
  const float* p = W + (size_t)(o_base + n) * ldk + kk + kb;
  const float4 f0 = ((const float4*)p)[0];
  const float4 f1 = ((const float4*)p)[1];
  const float4 f2 = ((const float4*)p)[2];
  const float4 f3 = ((const float4*)p)[3];
  v16h b;
  b[0]  = (_Float16)f0.x; b[1]  = (_Float16)f0.y;
  b[2]  = (_Float16)f0.z; b[3]  = (_Float16)f0.w;
  b[4]  = (_Float16)f1.x; b[5]  = (_Float16)f1.y;
  b[6]  = (_Float16)f1.z; b[7]  = (_Float16)f1.w;
  b[8]  = (_Float16)f2.x; b[9]  = (_Float16)f2.y;
  b[10] = (_Float16)f2.z; b[11] = (_Float16)f2.w;
  b[12] = (_Float16)f3.x; b[13] = (_Float16)f3.y;
  b[14] = (_Float16)f3.z; b[15] = (_Float16)f3.w;
  return b;
}

// ---------------------------------------------------------------------------
// Edge mean-aggregation: one wave32 per edge. Lanes read feat[src] as float4
// (coalesced 512B) and scatter with f32 atomics into sums[dst]. Count only
// depends on dst, so it's accumulated once (do_cnt=1 on the first pass).
__global__ void edge_agg_kernel(const float* __restrict__ feat,
                                const int* __restrict__ src,
                                const int* __restrict__ dst,
                                float* __restrict__ sums,
                                float* __restrict__ cnt,
                                int do_cnt) {
  const int wave = (blockIdx.x * blockDim.x + threadIdx.x) >> 5;
  const int lane = threadIdx.x & 31;
  const int s = src[wave];
  const int d = dst[wave];
  const float4 v = ((const float4*)(feat + (size_t)s * DF))[lane];
  float* o = sums + (size_t)d * DF + lane * 4;
  atomicAdd(o + 0, v.x);
  atomicAdd(o + 1, v.y);
  atomicAdd(o + 2, v.z);
  atomicAdd(o + 3, v.w);
  if (do_cnt && lane == 0) atomicAdd(cnt + d, 1.0f);
}

// ---------------------------------------------------------------------------
// Fused (optionally two-term) GEMM:
//   out = [relu]( A1*diag(1/max(cnt,1)) @ W1^T  (+ A2 @ W2^T)  + bias )
// Block = 256 threads (8 waves) covering 16 node rows; wave w owns output
// columns [16*w .. 16*w+15] and strides by 128 columns when O > 128.
// K = DF = 128 -> 4 chained v_wmma_f32_16x16x32_f16 per term per tile.
__global__ void gemm_wmma_kernel(const float* __restrict__ A1,
                                 const float* __restrict__ cnt,   // may be null
                                 const float* __restrict__ A2,    // may be null
                                 const float* __restrict__ W1,
                                 const float* __restrict__ W2,
                                 const float* __restrict__ bias,
                                 float* __restrict__ out,
                                 int O, int do_relu) {
  const int row_base = blockIdx.x * 16;
  const int wave = threadIdx.x >> 5;
  const int lane = threadIdx.x & 31;
  const int a_row = row_base + (lane & 15);   // A-fragment row for this lane
  const int hi = lane >> 4;

  float sc = 1.0f;
  if (cnt) sc = 1.0f / fmaxf(cnt[a_row], 1.0f);

  const float* a1p = A1 + (size_t)a_row * DF;
  const float* a2p = A2 ? (A2 + (size_t)a_row * DF) : nullptr;

  for (int o_base = wave * 16; o_base < O; o_base += 128) {
    const float bv = bias[o_base + (lane & 15)];
    v8f acc;
#pragma unroll
    for (int r = 0; r < 8; ++r) acc[r] = bv;

#pragma unroll
    for (int kk = 0; kk < DF; kk += 32) {
      v16h a = load_a_frag(a1p + kk, lane, sc);
      v16h b = load_b_frag(W1, DF, o_base, kk, lane);
      acc = __builtin_amdgcn_wmma_f32_16x16x32_f16(
          false, a, false, b, (short)0, acc, false, false);
    }
    if (a2p) {
#pragma unroll
      for (int kk = 0; kk < DF; kk += 32) {
        v16h a = load_a_frag(a2p + kk, lane, 1.0f);
        v16h b = load_b_frag(W2, DF, o_base, kk, lane);
        acc = __builtin_amdgcn_wmma_f32_16x16x32_f16(
            false, a, false, b, (short)0, acc, false, false);
      }
    }

    // D layout: VGPR r -> row M = r + 8*hi, column N = lane&15
    const int col = o_base + (lane & 15);
#pragma unroll
    for (int r = 0; r < 8; ++r) {
      float v = acc[r];
      if (do_relu) v = fmaxf(v, 0.0f);
      out[(size_t)(row_base + r + 8 * hi) * O + col] = v;
    }
  }
}

// ---------------------------------------------------------------------------
// Final decoder row-dot: out[n] = dot(h3[n, 0:256], W4) + b4. One wave/node.
__global__ void rowdot_kernel(const float* __restrict__ h3,
                              const float* __restrict__ W4,
                              const float* __restrict__ b4,
                              float* __restrict__ out) {
  const int node = (blockIdx.x * blockDim.x + threadIdx.x) >> 5;
  const int lane = threadIdx.x & 31;
  const float* row = h3 + (size_t)node * LH;
  float s = 0.0f;
#pragma unroll
  for (int i = lane; i < LH; i += 32) s += row[i] * W4[i];
#pragma unroll
  for (int off = 16; off > 0; off >>= 1) s += __shfl_xor(s, off, 32);
  if (lane == 0) out[node] = s + b4[0];
}

// ---------------------------------------------------------------------------
extern "C" void kernel_launch(void* const* d_in, const int* in_sizes, int n_in,
                              void* d_out, int out_size, void* d_ws, size_t ws_size,
                              hipStream_t stream) {
  const float* x   = (const float*)d_in[0];
  const int*   ei  = (const int*)d_in[1];      // (2, E), int32 per harness convention
  const float* W1l = (const float*)d_in[2];
  const float* b1  = (const float*)d_in[3];
  const float* W1r = (const float*)d_in[4];
  const float* W2l = (const float*)d_in[5];
  const float* b2  = (const float*)d_in[6];
  const float* W2r = (const float*)d_in[7];
  const float* W3  = (const float*)d_in[8];
  const float* b3  = (const float*)d_in[9];
  const float* W4  = (const float*)d_in[10];
  const float* b4  = (const float*)d_in[11];

  const int E = in_sizes[1] / 2;
  const int* src = ei;
  const int* dst = ei + E;

  // Workspace layout (floats):
  //   [0            ) sums : N*128
  //   [N*128        ) h1   : N*128
  //   [2*N*128      ) h2   : N*128
  //   [3*N*128      ) cnt  : N
  // h3 [N,256] reuses sums+h1 (both dead by decoder stage). ~154 MB total.
  float* sums = (float*)d_ws;
  float* h1   = sums + (size_t)NNODES * DF;
  float* h2   = h1   + (size_t)NNODES * DF;
  float* cnt  = h2   + (size_t)NNODES * DF;
  float* h3   = sums;   // N*256 == span of sums+h1

  const size_t featBytes = (size_t)NNODES * DF * sizeof(float);

  // conv1: mean-aggregate x, then h1 = relu(agg@W1l^T + b1 + x@W1r^T)
  hipMemsetAsync(sums, 0, featBytes, stream);
  hipMemsetAsync(cnt, 0, NNODES * sizeof(float), stream);
  edge_agg_kernel<<<E / 8, 256, 0, stream>>>(x, src, dst, sums, cnt, 1);
  gemm_wmma_kernel<<<NNODES / 16, 256, 0, stream>>>(
      sums, cnt, x, W1l, W1r, b1, h1, DF, 1);

  // conv2: mean-aggregate h1 (counts unchanged), h2 = agg@W2l^T + b2 + h1@W2r^T
  hipMemsetAsync(sums, 0, featBytes, stream);
  edge_agg_kernel<<<E / 8, 256, 0, stream>>>(h1, src, dst, sums, cnt, 0);
  gemm_wmma_kernel<<<NNODES / 16, 256, 0, stream>>>(
      sums, cnt, h1, W2l, W2r, b2, h2, DF, 0);

  // decoder: h3 = relu(h2@W3^T + b3) ; out = h3@W4^T + b4
  gemm_wmma_kernel<<<NNODES / 16, 256, 0, stream>>>(
      h2, nullptr, nullptr, W3, nullptr, b3, h3, LH, 1);
  rowdot_kernel<<<NNODES / 8, 256, 0, stream>>>(h3, W4, b4, (float*)d_out);
}
